// MambaGate2D_88201448390700
// MI455X (gfx1250) — matrix operand order repaired and checked by reading, not compiled, not run on hardware
//
#include <hip/hip_runtime.h>
#include <hip/hip_bf16.h>

// ---- problem constants (from reference) ----
#define D_MODEL 64
#define D_INNER 128
#define D_STATE 16
#define DT_RANK 4
#define D_CONVK 4
#define BATCH   16
#define LSEQ    4096               // 64*64 tokens per image
#define SEQS    32                 // 2*BATCH (fwd + reversed)
#define NTOK    (BATCH * LSEQ)     // 65536
#define NTOK2   (SEQS * LSEQ)      // 131072
#define XPROJ   (DT_RANK + 2 * D_STATE)  // 36

typedef __attribute__((ext_vector_type(16))) _Float16 v16h;
typedef __attribute__((ext_vector_type(8)))  float    v8f;

// ---------------------------------------------------------------------------
// K0: convert the three GEMM weight matrices to f16 (activations are O(1)
// post-LN; weights ~0.05 -> f16 is safe with f32 WMMA accumulation).
// ---------------------------------------------------------------------------
__global__ void mg_convert_weights(const float* __restrict__ wmi,
                                   const float* __restrict__ wxp,
                                   const float* __restrict__ wmo,
                                   _Float16* __restrict__ wmi16,
                                   _Float16* __restrict__ wxp16,
                                   _Float16* __restrict__ wmo16) {
    int i = blockIdx.x * blockDim.x + threadIdx.x;
    if (i < 2 * D_INNER * D_MODEL) wmi16[i] = (_Float16)wmi[i];
    if (i < XPROJ * D_INNER)       wxp16[i] = (_Float16)wxp[i];
    if (i < D_MODEL * D_INNER)     wmo16[i] = (_Float16)wmo[i];
}

// ---------------------------------------------------------------------------
// K1: in_proj 1x1 conv + LayerNorm; writes the f16 bidirectional token batch
// tk[32, L, 64]: rows 0..15 forward, rows 16..31 with L reversed.
// One thread per token; W_in staged in LDS (broadcast reads).
// ---------------------------------------------------------------------------
__global__ void mg_tok_ln(const float* __restrict__ x,
                          const float* __restrict__ W_in,
                          const float* __restrict__ b_in,
                          const float* __restrict__ ln_g,
                          const float* __restrict__ ln_b,
                          _Float16* __restrict__ tk) {
    __shared__ float sW[D_MODEL * D_MODEL];
    __shared__ float sb[D_MODEL], sg[D_MODEL], sbt[D_MODEL];
    int tid = threadIdx.x;
    for (int i = tid; i < D_MODEL * D_MODEL; i += blockDim.x) sW[i] = W_in[i];
    if (tid < D_MODEL) { sb[tid] = b_in[tid]; sg[tid] = ln_g[tid]; sbt[tid] = ln_b[tid]; }
    __syncthreads();

    int t = blockIdx.x * blockDim.x + tid;     // token id 0..65535
    int b = t >> 12;
    int l = t & (LSEQ - 1);
    const float* xp = x + ((size_t)b * D_MODEL) * LSEQ + l;   // x[b,c,l], stride L per c

    float acc[D_MODEL];
#pragma unroll
    for (int d = 0; d < D_MODEL; ++d) acc[d] = sb[d];
    for (int c = 0; c < D_MODEL; ++c) {
        float xv = xp[(size_t)c * LSEQ];
#pragma unroll
        for (int d = 0; d < D_MODEL; ++d) acc[d] += xv * sW[d * D_MODEL + c];
    }
    float mean = 0.f;
#pragma unroll
    for (int d = 0; d < D_MODEL; ++d) mean += acc[d];
    mean *= (1.f / D_MODEL);
    float var = 0.f;
#pragma unroll
    for (int d = 0; d < D_MODEL; ++d) { float c0 = acc[d] - mean; var += c0 * c0; }
    var *= (1.f / D_MODEL);
    float rstd = rsqrtf(var + 1e-5f);

    _Float16* pf = tk + ((size_t)b * LSEQ + l) * D_MODEL;
    _Float16* pb = tk + ((size_t)(BATCH + b) * LSEQ + (LSEQ - 1 - l)) * D_MODEL;
#pragma unroll
    for (int d = 0; d < D_MODEL; ++d) {
        float v = (acc[d] - mean) * rstd * sg[d] + sbt[d];
        _Float16 h = (_Float16)v;
        pf[d] = h;
        pb[d] = h;
    }
}

// ---------------------------------------------------------------------------
// WMMA GEMM: C[M,N](f32) = A[M,K](f16, row-major) * W[N,K](f16, row-major)^T
// One wave computes a 16 x (NT*16) output strip: A is loaded ONCE per k-step
// and reused across NT B-tiles (NT back-to-back v_wmma_f32_16x16x32_f16),
// cutting A traffic NT-fold and filling the XDL pipe between load clauses.
// VGPR layouts per CDNA5 ISA 7.12.2:
//   A (16x32 f16): lanes 0-15 row M=lane, halves = K {0..7,16..23};
//                  lanes 16-31 row M=lane-16, halves = K {8..15,24..31}.
//   B (32x16 f16): col N = lane&15; same K halves as A (lanes 0-15 K {0..7,16..23}, lanes 16-31 K {8..15,24..31}).
//   C/D (16x16 f32): VGPR r -> row r + 8*(lane>=16), col = lane&15.
// ---------------------------------------------------------------------------
template <int NT>
__global__ void mg_wmma_gemm(const _Float16* __restrict__ A,
                             const _Float16* __restrict__ W,
                             float* __restrict__ C,
                             int M, int N, int K, int groups_n) {
    int wave = blockIdx.x * (blockDim.x >> 5) + (threadIdx.x >> 5);
    int lane = threadIdx.x & 31;
    int tm = wave / groups_n;
    int tg = wave - tm * groups_n;
    int m0 = tm << 4;
    int n0 = tg * (NT * 16);
    if (m0 >= M) return;

    int r15 = lane & 15;
    int g   = lane >> 4;
    int arow = m0 + r15;
    int brow[NT];
#pragma unroll
    for (int j = 0; j < NT; ++j) {
        int r = n0 + j * 16 + r15;
        brow[j] = (r < N) ? r : (N - 1);   // clamp for edge tiles
    }

    v8f acc[NT];
#pragma unroll
    for (int j = 0; j < NT; ++j) acc[j] = (v8f){};

    for (int kk = 0; kk < K; kk += 32) {
        const _Float16* ap = A + (size_t)arow * K + kk;
        v16h a;
#pragma unroll
        for (int j = 0; j < 4; ++j) {
            int k0 = g * 8 + 2 * j;
            a[2 * j]     = ap[k0];
            a[2 * j + 1] = ap[k0 + 1];
            int k1 = 16 + g * 8 + 2 * j;
            a[8 + 2 * j]     = ap[k1];
            a[8 + 2 * j + 1] = ap[k1 + 1];
        }
        v16h b[NT];
#pragma unroll
        for (int j = 0; j < NT; ++j) {
            const _Float16* bp = W + (size_t)brow[j] * K + kk + g * 8;
#pragma unroll
            for (int e = 0; e < 16; ++e) b[j][e] = bp[(e < 8) ? e : e + 8];
        }
#pragma unroll
        for (int j = 0; j < NT; ++j)
            acc[j] = __builtin_amdgcn_wmma_f32_16x16x32_f16(false, a, false, b[j],
                                                            (short)0, acc[j], false, false);
    }
#pragma unroll
    for (int j = 0; j < NT; ++j) {
#pragma unroll
        for (int r = 0; r < 8; ++r) {
            int orow = m0 + r + g * 8;
            int ocol = n0 + j * 16 + r15;
            if (ocol < N) C[(size_t)orow * N + ocol] = acc[j][r];
        }
    }
}

// ---------------------------------------------------------------------------
// K3: causal depthwise conv (k=4) along L + bias + SiLU on the first 128
// columns of xz; channel-contiguous so lanes coalesce. Output xs as f16.
// ---------------------------------------------------------------------------
__global__ void mg_conv_silu(const float* __restrict__ xz,
                             const float* __restrict__ conv_w,
                             const float* __restrict__ conv_b,
                             _Float16* __restrict__ xs) {
    int ch    = threadIdx.x;            // 0..127
    int seq   = blockIdx.x >> 7;        // 0..31
    int chunk = blockIdx.x & 127;       // 128 chunks of 32 tokens
    int l0    = chunk * 32;

    float w0 = conv_w[ch * D_CONVK + 0];
    float w1 = conv_w[ch * D_CONVK + 1];
    float w2 = conv_w[ch * D_CONVK + 2];
    float w3 = conv_w[ch * D_CONVK + 3];
    float bb = conv_b[ch];

    const float* base = xz + ((size_t)seq * LSEQ) * (2 * D_INNER) + ch;
    for (int i = 0; i < 32; ++i) {
        int l = l0 + i;
        float acc = bb;
        if (l >= 3) acc += base[(size_t)(l - 3) * (2 * D_INNER)] * w0;
        if (l >= 2) acc += base[(size_t)(l - 2) * (2 * D_INNER)] * w1;
        if (l >= 1) acc += base[(size_t)(l - 1) * (2 * D_INNER)] * w2;
        acc += base[(size_t)l * (2 * D_INNER)] * w3;
        float s = acc / (1.f + __expf(-acc));          // SiLU
        xs[((size_t)seq * LSEQ + l) * D_INNER + ch] = (_Float16)s;
    }
}

// ---------------------------------------------------------------------------
// K5: selective scan. One block per sequence (32), one thread per channel
// (128). 16-element state in VGPRs; dt_proj(K=4)+softplus fused; B/C (36
// floats per step) staged in LDS; D-skip + SiLU(z) gate applied; f16 out.
// This is the latency-bound serial chain (4096 steps of exp+fma).
// ---------------------------------------------------------------------------
__global__ void mg_scan(const float* __restrict__ xdbl,
                        const _Float16* __restrict__ xs,
                        const float* __restrict__ xz,
                        const float* __restrict__ W_dt,
                        const float* __restrict__ b_dt,
                        const float* __restrict__ A_log,
                        const float* __restrict__ Dv,
                        _Float16* __restrict__ yg) {
    __shared__ float bc[XPROJ];
    int seq = blockIdx.x;
    int d   = threadIdx.x;

    float Arow[D_STATE];
#pragma unroll
    for (int n = 0; n < D_STATE; ++n) Arow[n] = -__expf(A_log[d * D_STATE + n]);
    float wd0 = W_dt[d * DT_RANK + 0];
    float wd1 = W_dt[d * DT_RANK + 1];
    float wd2 = W_dt[d * DT_RANK + 2];
    float wd3 = W_dt[d * DT_RANK + 3];
    float bdt = b_dt[d];
    float Dd  = Dv[d];

    float s[D_STATE];
#pragma unroll
    for (int n = 0; n < D_STATE; ++n) s[n] = 0.f;

    for (int l = 0; l < LSEQ; ++l) {
        __syncthreads();
        if (d < XPROJ) bc[d] = xdbl[((size_t)seq * LSEQ + l) * XPROJ + d];
        __syncthreads();

        float dtv = bdt + bc[0] * wd0 + bc[1] * wd1 + bc[2] * wd2 + bc[3] * wd3;
        float delta = (dtv > 20.f) ? dtv : log1pf(__expf(dtv));   // softplus
        float u  = (float)xs[((size_t)seq * LSEQ + l) * D_INNER + d];
        float du = delta * u;
        float y  = 0.f;
#pragma unroll
        for (int n = 0; n < D_STATE; ++n) {
            s[n] = __expf(delta * Arow[n]) * s[n] + du * bc[DT_RANK + n];
            y += s[n] * bc[DT_RANK + D_STATE + n];
        }
        y += u * Dd;
        float z = xz[((size_t)seq * LSEQ + l) * (2 * D_INNER) + D_INNER + d];
        float gate = z / (1.f + __expf(-z));
        yg[((size_t)seq * LSEQ + l) * D_INNER + d] = (_Float16)(y * gate);
    }
}

// ---------------------------------------------------------------------------
// K7: combine fwd/bwd halves, final 1x1 conv (W_out in LDS) + sigmoid-0.5,
// written in bchw layout (coalesced along l per output channel).
// ---------------------------------------------------------------------------
__global__ void mg_final(const float* __restrict__ ymo,
                         const float* __restrict__ W_out,
                         const float* __restrict__ b_out,
                         float* __restrict__ out) {
    __shared__ float sW[D_MODEL * D_MODEL];
    __shared__ float sb[D_MODEL];
    int tid = threadIdx.x;
    for (int i = tid; i < D_MODEL * D_MODEL; i += blockDim.x) sW[i] = W_out[i];
    if (tid < D_MODEL) sb[tid] = b_out[tid];
    __syncthreads();

    int t = blockIdx.x * blockDim.x + tid;   // 0..65535
    int b = t >> 12;
    int l = t & (LSEQ - 1);
    const float* pf = ymo + ((size_t)b * LSEQ + l) * D_MODEL;
    const float* pb = ymo + ((size_t)(BATCH + b) * LSEQ + (LSEQ - 1 - l)) * D_MODEL;

    float o[D_MODEL];
#pragma unroll
    for (int d = 0; d < D_MODEL; ++d) o[d] = 0.5f * (pf[d] + pb[d]);

    float* op = out + (size_t)b * D_MODEL * LSEQ + l;
    for (int c = 0; c < D_MODEL; ++c) {
        float acc = sb[c];
#pragma unroll
        for (int d = 0; d < D_MODEL; ++d) acc += o[d] * sW[c * D_MODEL + d];
        op[(size_t)c * LSEQ] = 1.f / (1.f + __expf(-acc)) - 0.5f;
    }
}

// ---------------------------------------------------------------------------
extern "C" void kernel_launch(void* const* d_in, const int* in_sizes, int n_in,
                              void* d_out, int out_size, void* d_ws, size_t ws_size,
                              hipStream_t stream) {
    const float* x      = (const float*)d_in[0];
    const float* W_in   = (const float*)d_in[1];
    const float* b_in   = (const float*)d_in[2];
    const float* ln_g   = (const float*)d_in[3];
    const float* ln_b   = (const float*)d_in[4];
    const float* W_mi   = (const float*)d_in[5];
    const float* conv_w = (const float*)d_in[6];
    const float* conv_b = (const float*)d_in[7];
    const float* W_xp   = (const float*)d_in[8];
    const float* W_dt   = (const float*)d_in[9];
    const float* b_dt   = (const float*)d_in[10];
    const float* A_log  = (const float*)d_in[11];
    const float* Dvec   = (const float*)d_in[12];
    const float* W_mo   = (const float*)d_in[13];
    const float* W_out  = (const float*)d_in[14];
    const float* b_out  = (const float*)d_in[15];
    float* out = (float*)d_out;

    // -------- workspace carve-up (aligned 256B) --------
    char* w = (char*)d_ws;
    size_t off = 0;
    auto alloc = [&](size_t bytes) -> void* {
        void* p = w + off;
        off = (off + bytes + 255) & ~(size_t)255;
        return p;
    };
    _Float16* tk16  = (_Float16*)alloc((size_t)NTOK2 * D_MODEL * sizeof(_Float16));      // 16.8 MB
    _Float16* wmi16 = (_Float16*)alloc((size_t)2 * D_INNER * D_MODEL * sizeof(_Float16));
    _Float16* wxp16 = (_Float16*)alloc((size_t)XPROJ * D_INNER * sizeof(_Float16));
    _Float16* wmo16 = (_Float16*)alloc((size_t)D_MODEL * D_INNER * sizeof(_Float16));
    float*    xz    = (float*)alloc((size_t)NTOK2 * 2 * D_INNER * sizeof(float));        // 134 MB
    _Float16* xs16  = (_Float16*)alloc((size_t)NTOK2 * D_INNER * sizeof(_Float16));      // 33.5 MB
    float*    xdbl  = (float*)alloc((size_t)NTOK2 * XPROJ * sizeof(float));              // 18.9 MB
    _Float16* y16   = (_Float16*)alloc((size_t)NTOK2 * D_INNER * sizeof(_Float16));      // 33.5 MB
    float*    ymo   = (float*)alloc((size_t)NTOK2 * D_MODEL * sizeof(float));            // 33.5 MB
    (void)ws_size; (void)in_sizes; (void)n_in; (void)out_size;

    // K0: weights -> f16
    mg_convert_weights<<<(2 * D_INNER * D_MODEL + 255) / 256, 256, 0, stream>>>(
        W_mi, W_xp, W_mo, wmi16, wxp16, wmo16);

    // K1: in_proj + LN -> bidirectional f16 tokens
    mg_tok_ln<<<NTOK / 256, 256, 0, stream>>>(x, W_in, b_in, ln_g, ln_b, tk16);

    // K2: xz = tk @ W_mi^T   (M=131072, N=256, K=64) — 16x64 strip per wave
    {
        int tiles_m = NTOK2 / 16;                 // 8192
        int groups_n = (2 * D_INNER) / 64;        // 4
        int waves = tiles_m * groups_n;
        mg_wmma_gemm<4><<<(waves + 7) / 8, 256, 0, stream>>>(
            tk16, wmi16, xz, NTOK2, 2 * D_INNER, D_MODEL, groups_n);
    }

    // K3: depthwise causal conv + SiLU -> xs (f16)
    mg_conv_silu<<<SEQS * 128, D_INNER, 0, stream>>>(xz, conv_w, conv_b, xs16);

    // K4: xdbl = xs @ W_xp^T (M=131072, N=36, K=128) — one wave covers all N
    {
        int tiles_m = NTOK2 / 16;                 // 8192
        int groups_n = 1;                         // NT=3 tiles cover 48 >= 36 cols
        int waves = tiles_m * groups_n;
        mg_wmma_gemm<3><<<(waves + 7) / 8, 256, 0, stream>>>(
            xs16, wxp16, xdbl, NTOK2, XPROJ, D_INNER, groups_n);
    }

    // K5: selective scan + D-skip + SiLU(z) gate -> y (f16)
    mg_scan<<<SEQS, D_INNER, 0, stream>>>(xdbl, xs16, xz, W_dt, b_dt, A_log, Dvec, y16);

    // K6: ymo = y @ W_mo^T (M=131072, N=64, K=128) — one wave covers all N
    {
        int tiles_m = NTOK2 / 16;                 // 8192
        int groups_n = 1;                         // NT=4 tiles cover N=64
        int waves = tiles_m * groups_n;
        mg_wmma_gemm<4><<<(waves + 7) / 8, 256, 0, stream>>>(
            y16, wmo16, ymo, NTOK2, D_MODEL, D_INNER, groups_n);
    }

    // K7: fwd/bwd combine + final 1x1 conv + sigmoid-0.5 (bchw)
    mg_final<<<NTOK / 256, 256, 0, stream>>>(ymo, W_out, b_out, out);
}
